// DRCLModule_3229815406975
// MI455X (gfx1250) — matrix operand order repaired and checked by reading, not compiled, not run on hardware
//
#include <hip/hip_runtime.h>
#include <math.h>

// ---------------------------------------------------------------------------
// MI455X (gfx1250) implementation. Dominant work: M=32768,K=2048,N=256 GEMM
// done with v_wmma_f32_16x16x32_bf16 (fp32 accumulate). BN stats are reduced
// from the WMMA accumulators; conv_b cancels through BatchNorm and is unused.
// Wave tiling: 2 M-subtiles x 8 N-subtiles per wave -> each B fragment feeds
// two independent WMMAs, and DS traffic is ~20 b128 loads per 16 WMMAs.
// All reductions are fixed-order (no float atomics) => deterministic output.
// ---------------------------------------------------------------------------

typedef __attribute__((ext_vector_type(16))) __bf16        v16bf;
typedef __attribute__((ext_vector_type(8)))  float         v8f;
typedef __attribute__((ext_vector_type(4)))  unsigned int  v4u;

#define NEG_INF (-1.0e9f)

// ---- workspace byte offsets ----
#define WS_FGS   0u         // 32768 f : fg scores
#define WS_BGS   131072u    // 32768 f : bg scores
#define WS_SUM   262144u    // 256 f   : column sums of z
#define WS_SQ    263168u    // 256 f   : column sums of z^2
#define WS_FGI   264192u    // 64 i
#define WS_BGI   264448u    // 64 i
#define WS_FGV   264704u    // 64 f
#define WS_BGV   264960u    // 64 f
#define WS_ZSEL  265216u    // 128*256 f : z rows of selected pixels
#define WS_PROJ  396288u    // 128*256 f : normalized projections
#define WS_MINV  527360u    // 512 f   : inverse norms of memory rows
#define WS_PART  529408u    // 256 blocks * 512 f : per-block partial stats

__device__ inline unsigned short f2bf(float x) {   // fp32 -> bf16, RNE
  unsigned int u = __float_as_uint(x);
  u += 0x7FFFu + ((u >> 16) & 1u);
  return (unsigned short)(u >> 16);
}

// ---------------------------------------------------------------------------
// Kernel 1: reliability / difficulty masks -> fg/bg mining scores
// ---------------------------------------------------------------------------
__global__ void k_scores(const float* __restrict__ po, const float* __restrict__ pa,
                         const float* __restrict__ unc, const int* __restrict__ lab,
                         float* __restrict__ fgs, float* __restrict__ bgs) {
  int i = blockIdx.x * blockDim.x + threadIdx.x;
  if (i >= 32768) return;
  int b = i >> 12, p = i & 4095;
  int ob = b * 2 * 4096;
  bool co = po[ob + 4096 + p] > po[ob + p];   // argmax over 2 ch (tie -> 0)
  bool ca = pa[ob + 4096 + p] > pa[ob + p];
  float u = unc[i];
  bool diff = (co == ca) && (u > 0.5f);
  int l = lab[i];
  fgs[i] = (diff && l == 1) ? u : NEG_INF;
  bgs[i] = (diff && l == 0) ? u : NEG_INF;
}

// ---------------------------------------------------------------------------
// Kernel 2: iterative top-64 (value desc, lowest index on ties == jax top_k)
// ---------------------------------------------------------------------------
__global__ void k_topk(float* __restrict__ fgs, float* __restrict__ bgs,
                       int* __restrict__ fgi, int* __restrict__ bgi,
                       float* __restrict__ fgv, float* __restrict__ bgv) {
  __shared__ float sv[256];
  __shared__ int   si[256];
  int t = threadIdx.x;
  for (int pass = 0; pass < 2; ++pass) {
    float* sc = pass ? bgs : fgs;
    int*   oi = pass ? bgi : fgi;
    float* ov = pass ? bgv : fgv;
    for (int r = 0; r < 64; ++r) {
      float bv = -3.0e9f; int bi = 1 << 30;
      for (int i = t; i < 32768; i += 256) {
        float x = sc[i];
        if (x > bv || (x == bv && i < bi)) { bv = x; bi = i; }
      }
      sv[t] = bv; si[t] = bi;
      __syncthreads();
      for (int s = 128; s > 0; s >>= 1) {
        if (t < s) {
          float xv = sv[t + s]; int xi = si[t + s];
          if (xv > sv[t] || (xv == sv[t] && xi < si[t])) { sv[t] = xv; si[t] = xi; }
        }
        __syncthreads();
      }
      if (t == 0) {
        int w = si[0]; float v = sv[0];
        oi[r] = w;
        ov[r] = (v > NEG_INF * 0.5f) ? 1.0f : 0.0f;
        sc[w] = -2.0e9f;                 // knock out; stays above init sentinel
      }
      __threadfence();
      __syncthreads();
    }
  }
}

// ---------------------------------------------------------------------------
// Shared WMMA helpers. LDS layouts chosen so each lane's bf16 fragment is two
// contiguous 16B chunks (ds_load_b128):
//   A tile : lA[m][k], m in [0,128), k in [0,32)
//   B tile : lB[n][k]  (K-transposed), n in [0,256), k in [0,32)
// ISA 16-bit A 16x32 map: lanes0-15 VGPR0-3 = K0..7, VGPR4-7 = K16..23;
// lanes16-31: K8..15 / K24..31.  B 32x16: lanes0-15 col=lane K0..7 / K16..23,
// lanes16-31 col=lane-16 K8..15 / K24..31.
// Wave w owns M-subtiles {2*(w&3), 2*(w&3)+1} and N-subtiles (w>>2)*8..+8.
// ---------------------------------------------------------------------------
__device__ inline void stageB(const float* __restrict__ w, int kc,
                              unsigned short* lB, int t) {
  for (int it = t; it < 1024; it += 256) {
    int k = (it & 15) * 2, n0 = (it >> 4) * 4;
    float4 f0 = *(const float4*)(w + (size_t)(kc + k) * 256 + n0);
    float4 f1 = *(const float4*)(w + (size_t)(kc + k + 1) * 256 + n0);
    lB[(n0 + 0) * 32 + k] = f2bf(f0.x); lB[(n0 + 0) * 32 + k + 1] = f2bf(f1.x);
    lB[(n0 + 1) * 32 + k] = f2bf(f0.y); lB[(n0 + 1) * 32 + k + 1] = f2bf(f1.y);
    lB[(n0 + 2) * 32 + k] = f2bf(f0.z); lB[(n0 + 2) * 32 + k + 1] = f2bf(f1.z);
    lB[(n0 + 3) * 32 + k] = f2bf(f0.w); lB[(n0 + 3) * 32 + k + 1] = f2bf(f1.w);
  }
}

__device__ inline void wmma_step(const unsigned short* lA, const unsigned short* lB,
                                 int mpair, int nhalf, int l15, int half,
                                 v8f* acc0, v8f* acc1) {
  union { v4u u[2]; v16bf v; } a0, a1, bfr;
  const unsigned short* ab0 = lA + ((mpair * 2) * 16 + l15) * 32;
  const unsigned short* ab1 = ab0 + 16 * 32;
  a0.u[0] = *(const v4u*)(ab0 + half * 8);        // K 0..7  / 8..15
  a0.u[1] = *(const v4u*)(ab0 + 16 + half * 8);   // K 16..23 / 24..31
  a1.u[0] = *(const v4u*)(ab1 + half * 8);
  a1.u[1] = *(const v4u*)(ab1 + 16 + half * 8);
#pragma unroll
  for (int j = 0; j < 8; ++j) {
    const unsigned short* bb = lB + ((nhalf * 8 + j) * 16 + l15) * 32 + half * 8;
    bfr.u[0] = *(const v4u*)(bb);
    bfr.u[1] = *(const v4u*)(bb + 16);
    acc0[j] = __builtin_amdgcn_wmma_f32_16x16x32_bf16(
        false, a0.v, false, bfr.v, (short)0, acc0[j], false, false);
    acc1[j] = __builtin_amdgcn_wmma_f32_16x16x32_bf16(
        false, a1.v, false, bfr.v, (short)0, acc1[j], false, false);
  }
}

// ---------------------------------------------------------------------------
// Kernel 3: big GEMM, reduced to per-channel sum / sumsq (BN statistics).
// Grid: 256 blocks x 256 threads (8 wave32). Each block: 128 pixels x 256 out.
// ---------------------------------------------------------------------------
__global__ void __launch_bounds__(256)
k_gemm_stats(const float* __restrict__ feat, const float* __restrict__ w,
             float* __restrict__ part) {
  __shared__ __align__(16) unsigned short lA[128 * 32];
  __shared__ __align__(16) unsigned short lB[256 * 32];
  __shared__ float lpS[8 * 256];
  __shared__ float lpQ[8 * 256];
  int t = threadIdx.x;
  int lane = t & 31, half = lane >> 4, l15 = lane & 15, wv = t >> 5;
  int mpair = wv & 3, nhalf = wv >> 2;
  int gm = blockIdx.x * 128;                 // tile of 128 consecutive pixels
  int b = gm >> 12, p0 = gm & 4095;          // 128 | 4096 -> stays in batch b
  const size_t fb = (size_t)b * 2048 * 4096 + (size_t)p0;

  v8f acc0[8], acc1[8];
  v8f zero = {0.f, 0.f, 0.f, 0.f, 0.f, 0.f, 0.f, 0.f};
#pragma unroll
  for (int n = 0; n < 8; ++n) { acc0[n] = zero; acc1[n] = zero; }

  for (int kc = 0; kc < 2048; kc += 32) {
    // stage A: 32 channels x 128 contiguous pixels, fp32 -> bf16, transpose
    for (int it = t; it < 512; it += 256) {
      int k = (it & 15) * 2, m0 = (it >> 4) * 4;
      float4 f0 = *(const float4*)(feat + fb + (size_t)(kc + k) * 4096 + m0);
      float4 f1 = *(const float4*)(feat + fb + (size_t)(kc + k + 1) * 4096 + m0);
      lA[(m0 + 0) * 32 + k] = f2bf(f0.x); lA[(m0 + 0) * 32 + k + 1] = f2bf(f1.x);
      lA[(m0 + 1) * 32 + k] = f2bf(f0.y); lA[(m0 + 1) * 32 + k + 1] = f2bf(f1.y);
      lA[(m0 + 2) * 32 + k] = f2bf(f0.z); lA[(m0 + 2) * 32 + k + 1] = f2bf(f1.z);
      lA[(m0 + 3) * 32 + k] = f2bf(f0.w); lA[(m0 + 3) * 32 + k + 1] = f2bf(f1.w);
    }
    stageB(w, kc, lB, t);
    __syncthreads();
    wmma_step(lA, lB, mpair, nhalf, l15, half, acc0, acc1);
    __syncthreads();
  }

  // reduce accumulators -> per-column sum / sumsq over this wave's 32 rows
#pragma unroll
  for (int j = 0; j < 8; ++j) {
    union { v8f v; float f[8]; } a0, a1;
    a0.v = acc0[j]; a1.v = acc1[j];
    float s = 0.f, q = 0.f;
#pragma unroll
    for (int r = 0; r < 8; ++r) {
      s += a0.f[r] + a1.f[r];
      q += a0.f[r] * a0.f[r] + a1.f[r] * a1.f[r];
    }
    s += __shfl_xor(s, 16, 32);              // combine M 0..7 with M 8..15
    q += __shfl_xor(q, 16, 32);
    if (half == 0) {
      lpS[wv * 256 + (nhalf * 8 + j) * 16 + l15] = s;
      lpQ[wv * 256 + (nhalf * 8 + j) * 16 + l15] = q;
    }
  }
  __syncthreads();
  if (t < 256) {                             // fixed-order fold over mpairs
    int w0 = (t < 128) ? 0 : 4;              // waves 0-3 own cols 0..127
    float s = 0.f, q = 0.f;
    for (int k = 0; k < 4; ++k) { s += lpS[(w0 + k) * 256 + t]; q += lpQ[(w0 + k) * 256 + t]; }
    part[(size_t)blockIdx.x * 512 + t] = s;
    part[(size_t)blockIdx.x * 512 + 256 + t] = q;
  }
}

// Deterministic fold of per-block partials (fixed order over 256 blocks).
__global__ void k_reduce(const float* __restrict__ part,
                         float* __restrict__ gsum, float* __restrict__ gsq) {
  int t = threadIdx.x;   // 512 threads
  float s = 0.f;
  for (int blk = 0; blk < 256; ++blk) s += part[(size_t)blk * 512 + t];
  if (t < 256) gsum[t] = s; else gsq[t - 256] = s;
}

// ---------------------------------------------------------------------------
// Kernel 4: gather-GEMM for the 128 mined pixels (z rows). 1 block, 8 waves.
// ---------------------------------------------------------------------------
__global__ void __launch_bounds__(256)
k_gemm_sel(const float* __restrict__ feat, const float* __restrict__ w,
           const int* __restrict__ fgi, const int* __restrict__ bgi,
           float* __restrict__ zsel) {
  __shared__ __align__(16) unsigned short lA[128 * 32];
  __shared__ __align__(16) unsigned short lB[256 * 32];
  __shared__ int sel[128];
  int t = threadIdx.x;
  int lane = t & 31, half = lane >> 4, l15 = lane & 15, wv = t >> 5;
  int mpair = wv & 3, nhalf = wv >> 2;
  if (t < 128) sel[t] = (t < 64) ? fgi[t] : bgi[t - 64];
  __syncthreads();

  v8f acc0[8], acc1[8];
  v8f zero = {0.f, 0.f, 0.f, 0.f, 0.f, 0.f, 0.f, 0.f};
#pragma unroll
  for (int n = 0; n < 8; ++n) { acc0[n] = zero; acc1[n] = zero; }

  for (int kc = 0; kc < 2048; kc += 32) {
    for (int it = t; it < 2048; it += 256) {     // gather 128 rows x 32 k
      int m = it >> 4, k = (it & 15) * 2;
      int g = sel[m];
      size_t base = ((size_t)((g >> 12) * 2048 + kc + k)) * 4096 + (size_t)(g & 4095);
      lA[m * 32 + k]     = f2bf(feat[base]);
      lA[m * 32 + k + 1] = f2bf(feat[base + 4096]);
    }
    stageB(w, kc, lB, t);
    __syncthreads();
    wmma_step(lA, lB, mpair, nhalf, l15, half, acc0, acc1);
    __syncthreads();
  }

#pragma unroll
  for (int j = 0; j < 8; ++j) {
    union { v8f v; float f[8]; } a0, a1;
    a0.v = acc0[j]; a1.v = acc1[j];
    int col = (nhalf * 8 + j) * 16 + l15;
#pragma unroll
    for (int r = 0; r < 8; ++r) {
      int row0 = (mpair * 2) * 16 + half * 8 + r;
      zsel[row0 * 256 + col] = a0.f[r];
      zsel[(row0 + 16) * 256 + col] = a1.f[r];
    }
  }
}

// ---------------------------------------------------------------------------
// Kernel 5: BN finalize + ReLU + L2-normalize + 4 InfoNCE heads -> scalar.
// ---------------------------------------------------------------------------
__global__ void __launch_bounds__(256)
k_final(const float* __restrict__ gsum, const float* __restrict__ gsq,
        const float* __restrict__ gamma, const float* __restrict__ beta,
        const float* __restrict__ zsel, const float* __restrict__ mpos,
        const float* __restrict__ mneg, const float* __restrict__ fgv,
        const float* __restrict__ bgv, float* __restrict__ projn,
        float* __restrict__ minv, float* __restrict__ out) {
  __shared__ float mu[256], istd[256], ga[256], be[256], lres[256];
  int t = threadIdx.x;
  {
    float m = gsum[t] * (1.0f / 32768.0f);
    float v = gsq[t] * (1.0f / 32768.0f) - m * m;
    mu[t] = m; istd[t] = rsqrtf(v + 1e-5f);
    ga[t] = gamma[t]; be[t] = beta[t];
    lres[t] = 0.f;
  }
  __syncthreads();
  if (t < 128) {                        // BN + ReLU + L2 normalize one row
    float ss = 0.f;
    for (int d = 0; d < 256; ++d) {
      float val = ga[d] * ((zsel[t * 256 + d] - mu[d]) * istd[d]) + be[d];
      val = fmaxf(val, 0.f);
      projn[t * 256 + d] = val;
      ss += val * val;
    }
    float inv = 1.0f / (sqrtf(ss) + 1e-12f);
    for (int d = 0; d < 256; ++d) projn[t * 256 + d] *= inv;
  }
  for (int i = t; i < 512; i += 256) {  // memory-bank inverse row norms
    const float* row = (i < 256) ? (mpos + i * 256) : (mneg + (i - 256) * 256);
    float ss = 0.f;
    for (int d = 0; d < 256; ++d) ss += row[d] * row[d];
    minv[i] = 1.0f / (sqrtf(ss) + 1e-12f);
  }
  __threadfence();
  __syncthreads();
  if (t < 64) {                         // 4 heads x 16 anchors
    int c = t >> 4, a = t & 15;
    bool bgq = (c == 1) || (c == 3);
    const float* q = projn + (bgq ? (64 + a) : a) * 256;
    float qv = bgq ? bgv[a] : fgv[a];
    float pe = 0.f, ne = 0.f;
    if (c < 2) {                        // local: fg/bg pools of 64
      for (int p = 0; p < 64; ++p) {
        const float* pr = projn + ((c == 0) ? p : (64 + p)) * 256;
        const float* nr = projn + ((c == 0) ? (64 + p) : p) * 256;
        float dp = 0.f, dn = 0.f;
        for (int d = 0; d < 256; ++d) { dp += q[d] * pr[d]; dn += q[d] * nr[d]; }
        float pv = (c == 0) ? fgv[p] : bgv[p];
        float nv = (c == 0) ? bgv[p] : fgv[p];
        pe += expf(dp * 10.0f) * pv;
        ne += expf(dn * 10.0f) * nv;
      }
    } else {                            // global: memory banks of 256
      for (int p = 0; p < 256; ++p) {
        const float* pr = ((c == 2) ? mpos : mneg) + p * 256;
        const float* nr = ((c == 2) ? mneg : mpos) + p * 256;
        float ip = (c == 2) ? minv[p] : minv[256 + p];
        float in_ = (c == 2) ? minv[256 + p] : minv[p];
        float dp = 0.f, dn = 0.f;
        for (int d = 0; d < 256; ++d) { dp += q[d] * pr[d]; dn += q[d] * nr[d]; }
        pe += expf(dp * ip * 10.0f);
        ne += expf(dn * in_ * 10.0f);
      }
    }
    float loss = -logf(pe / (pe + ne + 1e-8f) + 1e-8f);
    lres[t] = loss * qv;
  }
  __syncthreads();
  if (t == 0) {
    float lsum = 0.f;
    for (int i = 0; i < 64; ++i) lsum += lres[i];
    float c12 = 0.f;
    for (int a = 0; a < 16; ++a) c12 += fgv[a] + bgv[a];
    out[0] = lsum / fmaxf(c12, 1.0f);   // (l1+l2)/n + (g1+g2)/n
  }
}

// ---------------------------------------------------------------------------
extern "C" void kernel_launch(void* const* d_in, const int* in_sizes, int n_in,
                              void* d_out, int out_size, void* d_ws, size_t ws_size,
                              hipStream_t stream) {
  (void)in_sizes; (void)n_in; (void)out_size; (void)ws_size;
  const float* feat = (const float*)d_in[0];
  const float* po   = (const float*)d_in[1];
  const float* pa   = (const float*)d_in[2];
  const float* unc  = (const float*)d_in[3];
  const int*   lab  = (const int*)d_in[4];
  const float* cw   = (const float*)d_in[5];
  // d_in[6] = conv_b: cancels exactly through BatchNorm -> unused.
  const float* gam  = (const float*)d_in[7];
  const float* bet  = (const float*)d_in[8];
  const float* mpos = (const float*)d_in[9];
  const float* mneg = (const float*)d_in[10];

  char* ws = (char*)d_ws;
  float* fgs  = (float*)(ws + WS_FGS);
  float* bgs  = (float*)(ws + WS_BGS);
  float* gsum = (float*)(ws + WS_SUM);
  float* gsq  = (float*)(ws + WS_SQ);
  int*   fgi  = (int*)(ws + WS_FGI);
  int*   bgi  = (int*)(ws + WS_BGI);
  float* fgv  = (float*)(ws + WS_FGV);
  float* bgv  = (float*)(ws + WS_BGV);
  float* zsel = (float*)(ws + WS_ZSEL);
  float* prj  = (float*)(ws + WS_PROJ);
  float* minv = (float*)(ws + WS_MINV);
  float* part = (float*)(ws + WS_PART);

  k_scores<<<128, 256, 0, stream>>>(po, pa, unc, lab, fgs, bgs);
  k_topk<<<1, 256, 0, stream>>>(fgs, bgs, fgi, bgi, fgv, bgv);
  k_gemm_stats<<<256, 256, 0, stream>>>(feat, cw, part);
  k_reduce<<<1, 512, 0, stream>>>(part, gsum, gsq);
  k_gemm_sel<<<1, 256, 0, stream>>>(feat, cw, fgi, bgi, zsel);
  k_final<<<1, 256, 0, stream>>>(gsum, gsq, gam, bet, zsel, mpos, mneg,
                                 fgv, bgv, prj, minv, (float*)d_out);
}
